// ModuleBlock_1709396984052
// MI455X (gfx1250) — matrix operand order repaired and checked by reading, not compiled, not run on hardware
//
#include <hip/hip_runtime.h>

// ---------------- problem constants ----------------
#define B_    2
#define S_    2048
#define H_    2048
#define NH_   16
#define HD_   128
#define FF_   8192
#define NROW  (B_ * S_)               // 4096 rows
#define EPS_  1e-5f
#define ATTN_SCALE 0.08838834764831845f   // HD^-0.5
#define ROPE_C 0.14391156813f             // ln(10000)/64

typedef __bf16 bf16_t;
typedef __bf16 v16bf __attribute__((ext_vector_type(16)));
typedef float  v8f   __attribute__((ext_vector_type(8)));
typedef unsigned int u32x4 __attribute__((ext_vector_type(4)));  // 16-byte mover

union ABu  { v16bf v; u32x4 q[2]; };     // 32B fragment assembled from two 16B chunks
union U4B  { u32x4 q; bf16_t e[8]; };    // unpack 8 bf16 from 16B

// ---- CDNA5 async global->LDS DMA path (ASYNCcnt), guarded for portability ---
#if __has_builtin(__builtin_amdgcn_global_load_async_to_lds_b128) && \
    __has_builtin(__builtin_amdgcn_s_wait_asynccnt)
#define USE_ASYNC_LDS 1
#else
#define USE_ASYNC_LDS 0
#endif

#if USE_ASYNC_LDS
// builtin signature (from hipcc diagnostic): (v4i AS1*, v4i AS3*, imm offset, imm cpol)
typedef __attribute__((vector_size(16))) int v4i_t;
typedef __attribute__((address_space(1))) v4i_t* gptr_t;
typedef __attribute__((address_space(3))) v4i_t* lptr_t;
static __device__ inline void async_cp16(const bf16_t* g, bf16_t* l) {
  // 16 bytes: memory -> LDS directly, no VGPR round-trip; tracked by ASYNCcnt.
  __builtin_amdgcn_global_load_async_to_lds_b128((gptr_t)g, (lptr_t)l, 0, 0);
}
#define GEMM_NBUF 2
#else
#define GEMM_NBUF 1
#endif

static __device__ inline v8f wmma_bf16(v16bf a, v16bf b, v8f c) {
  // D(16x16,f32) = A(16x32,bf16) * B(32x16,bf16) + C
  return __builtin_amdgcn_wmma_f32_16x16x32_bf16(false, a, false, b, (short)0, c,
                                                 false, false);
}

// ---------------- block-wide f32 sum (256 threads, 8 waves) ----------------
static __device__ inline float blk_sum(float v) {
  __shared__ float red[8];
  const int lane = threadIdx.x & 31;
  const int wave = threadIdx.x >> 5;
#pragma unroll
  for (int d = 16; d >= 1; d >>= 1) v += __shfl_xor(v, d, 32);
  if (lane == 0) red[wave] = v;
  __syncthreads();
  float s = red[0] + red[1] + red[2] + red[3] + red[4] + red[5] + red[6] + red[7];
  __syncthreads();   // allow safe reuse
  return s;
}

// ---------------- generic C = A(MxK) * W(NxK)^T, bf16 in, f32 or bf16 out ----
// block tile 128x128, 256 threads = 8 waves in a 2(M) x 4(N) grid; wave = 64x32.
// Async build: double-buffered LDS, one barrier + s_wait_asynccnt per K-slab.
__global__ __launch_bounds__(256) void gemm_bt_kernel(
    const bf16_t* __restrict__ A, const bf16_t* __restrict__ W,
    void* __restrict__ Cout, int M, int Nc, int K, int store_bf16)
{
  __shared__ bf16_t sA[GEMM_NBUF][128 * 32];
  __shared__ bf16_t sB[GEMM_NBUF][128 * 32];
  const int tid  = threadIdx.x;
  const int lane = tid & 31, wave = tid >> 5;
  const int lh = lane >> 4, ll = lane & 15;
  const int wm = wave >> 2, wn = wave & 3;
  const long bm = (long)blockIdx.y * 128;
  const long bn = (long)blockIdx.x * 128;

  v8f acc[4][2];
#pragma unroll
  for (int i = 0; i < 4; i++)
#pragma unroll
    for (int j = 0; j < 2; j++) acc[i][j] = (v8f){0.f,0.f,0.f,0.f,0.f,0.f,0.f,0.f};

  const int lrow = tid >> 1;            // 0..127
  const int lcol = (tid & 1) * 16;      // 0 / 16

#if USE_ASYNC_LDS
  auto stage = [&](int buf, int kb) {
    const long aoff = (bm + lrow) * (long)K + kb + lcol;
    const long boff = (bn + lrow) * (long)K + kb + lcol;
    async_cp16(&A[aoff],     &sA[buf][lrow * 32 + lcol]);
    async_cp16(&A[aoff + 8], &sA[buf][lrow * 32 + lcol + 8]);
    async_cp16(&W[boff],     &sB[buf][lrow * 32 + lcol]);
    async_cp16(&W[boff + 8], &sB[buf][lrow * 32 + lcol + 8]);
  };
  stage(0, 0);                           // prologue prefetch
  int buf = 0;
#else
  const int buf = 0;
#endif

  for (int kb = 0; kb < K; kb += 32) {
#if USE_ASYNC_LDS
    __builtin_amdgcn_s_wait_asynccnt(0);   // my slab writes have landed in LDS
    __syncthreads();                       // everyone's writes landed, prev reads done
    if (kb + 32 < K) stage(buf ^ 1, kb + 32);   // prefetch next slab into alt buffer
#else
    const long aoff = (bm + lrow) * (long)K + kb + lcol;
    const long boff = (bn + lrow) * (long)K + kb + lcol;
    *(u32x4*)&sA[0][lrow * 32 + lcol]     = *(const u32x4*)&A[aoff];
    *(u32x4*)&sA[0][lrow * 32 + lcol + 8] = *(const u32x4*)&A[aoff + 8];
    *(u32x4*)&sB[0][lrow * 32 + lcol]     = *(const u32x4*)&W[boff];
    *(u32x4*)&sB[0][lrow * 32 + lcol + 8] = *(const u32x4*)&W[boff + 8];
    __syncthreads();
#endif

    v16bf aF[4], bF[2];
#pragma unroll
    for (int i = 0; i < 4; i++) {
      const int r = wm * 64 + i * 16 + ll;           // A row
      ABu u;
      u.q[0] = *(const u32x4*)&sA[buf][r * 32 + lh * 8];       // K = lh*8 .. +7
      u.q[1] = *(const u32x4*)&sA[buf][r * 32 + lh * 8 + 16];  // K = lh*8+16 .. +23
      aF[i] = u.v;
    }
#pragma unroll
    for (int j = 0; j < 2; j++) {
      const int r = wn * 32 + j * 16 + ll;           // W row = output col
      ABu ub;
      ub.q[0] = *(const u32x4*)&sB[buf][r * 32 + lh * 8];       // K = lh*8 .. +7
      ub.q[1] = *(const u32x4*)&sB[buf][r * 32 + lh * 8 + 16];  // K = lh*8+16 .. +23
      bF[j] = ub.v;
    }
#pragma unroll
    for (int i = 0; i < 4; i++)
#pragma unroll
      for (int j = 0; j < 2; j++)
        acc[i][j] = wmma_bf16(aF[i], bF[j], acc[i][j]);

#if USE_ASYNC_LDS
    buf ^= 1;
#else
    __syncthreads();
#endif
  }

#pragma unroll
  for (int i = 0; i < 4; i++)
#pragma unroll
    for (int j = 0; j < 2; j++)
#pragma unroll
      for (int v = 0; v < 8; v++) {
        const long m = bm + wm * 64 + i * 16 + v + 8 * lh;
        const long n = bn + wn * 32 + j * 16 + ll;
        const float val = acc[i][j][v];
        if (store_bf16) ((bf16_t*)Cout)[m * (long)Nc + n] = (bf16_t)val;
        else            ((float*) Cout)[m * (long)Nc + n] = val;
      }
}

// ---------------- flash attention, bf16 WMMA, online softmax -----------------
// grid: (S/128, NH, B); 256 threads = 8 waves, wave w owns 16 query rows.
__global__ __launch_bounds__(256) void attn_kernel(
    const bf16_t* __restrict__ Q, const bf16_t* __restrict__ Kb,
    const bf16_t* __restrict__ Vb, const float* __restrict__ mask,
    bf16_t* __restrict__ O)
{
  __shared__ bf16_t sK[32 * 128];       // [key][hd]
  __shared__ bf16_t sVt[128 * 32];      // [hd][key]  (transposed)
  __shared__ bf16_t sP[8][16 * 32];     // per-wave P tile

  const int bb = blockIdx.z, h = blockIdx.y, qt = blockIdx.x;
  const int tid = threadIdx.x, wave = tid >> 5, lane = tid & 31;
  const int lh = lane >> 4, ll = lane & 15;
  const int qbase = qt * 128 + wave * 16;

  // Q fragments for this wave's 16 rows, HD=128 -> 4 K-chunks of 32
  v16bf qf[4];
  {
    const long qoff = ((long)(bb * S_) + qbase + ll) * H_ + h * HD_;
#pragma unroll
    for (int kk = 0; kk < 4; kk++) {
      ABu u;
      u.q[0] = *(const u32x4*)&Q[qoff + kk * 32 + lh * 8];
      u.q[1] = *(const u32x4*)&Q[qoff + kk * 32 + lh * 8 + 16];
      qf[kk] = u.v;
    }
  }

  float mrow[8], lsum[8];
  v8f accO[8];
#pragma unroll
  for (int v = 0; v < 8; v++) { mrow[v] = -3.0e38f; lsum[v] = 0.f; }
#pragma unroll
  for (int n = 0; n < 8; n++) accO[n] = (v8f){0.f,0.f,0.f,0.f,0.f,0.f,0.f,0.f};

  const int krow = tid >> 3;            // 0..31 (key within tile)
  const int kcol = (tid & 7) * 16;      // 0..112 (hd)

  for (int kt = 0; kt < S_; kt += 32) {
    // stage K tile (async DMA when available) and transposed V tile
    const long koff = ((long)(bb * S_) + kt + krow) * H_ + h * HD_ + kcol;
#if USE_ASYNC_LDS
    async_cp16(&Kb[koff],     &sK[krow * 128 + kcol]);
    async_cp16(&Kb[koff + 8], &sK[krow * 128 + kcol + 8]);
#else
    *(u32x4*)&sK[krow * 128 + kcol]     = *(const u32x4*)&Kb[koff];
    *(u32x4*)&sK[krow * 128 + kcol + 8] = *(const u32x4*)&Kb[koff + 8];
#endif
    U4B va, vb2;
    va.q  = *(const u32x4*)&Vb[koff];
    vb2.q = *(const u32x4*)&Vb[koff + 8];
#pragma unroll
    for (int j = 0; j < 8; j++) sVt[(kcol + j) * 32 + krow]     = va.e[j];
#pragma unroll
    for (int j = 0; j < 8; j++) sVt[(kcol + 8 + j) * 32 + krow] = vb2.e[j];
#if USE_ASYNC_LDS
    __builtin_amdgcn_s_wait_asynccnt(0);
#endif
    __syncthreads();

    // scores: S = Q * K^T  -> two 16x16 tiles per wave
    v8f sc[2];
    sc[0] = (v8f){0.f,0.f,0.f,0.f,0.f,0.f,0.f,0.f};
    sc[1] = (v8f){0.f,0.f,0.f,0.f,0.f,0.f,0.f,0.f};
#pragma unroll
    for (int nt = 0; nt < 2; nt++) {
      const int key = nt * 16 + ll;
#pragma unroll
      for (int kk = 0; kk < 4; kk++) {
        ABu uk;
        uk.q[0] = *(const u32x4*)&sK[key * 128 + kk * 32 + lh * 8];
        uk.q[1] = *(const u32x4*)&sK[key * 128 + kk * 32 + lh * 8 + 16];
        v16bf bK = uk.v;
        sc[nt] = wmma_bf16(qf[kk], bK, sc[nt]);
      }
    }

    // online softmax update (rows are (v, lane-half); cols across 16 lanes)
#pragma unroll
    for (int v = 0; v < 8; v++) {
      const long qrow = qbase + v + 8 * lh;
      float s0 = sc[0][v] * ATTN_SCALE + mask[qrow * (long)S_ + kt + ll];
      float s1 = sc[1][v] * ATTN_SCALE + mask[qrow * (long)S_ + kt + 16 + ll];
      float mx = fmaxf(s0, s1);
#pragma unroll
      for (int d = 8; d >= 1; d >>= 1) mx = fmaxf(mx, __shfl_xor(mx, d, 32));
      const float mnew = fmaxf(mrow[v], mx);
      const float corr = __expf(mrow[v] - mnew);
      const float p0 = __expf(s0 - mnew);
      const float p1 = __expf(s1 - mnew);
      float ps = p0 + p1;
#pragma unroll
      for (int d = 8; d >= 1; d >>= 1) ps += __shfl_xor(ps, d, 32);
      lsum[v] = lsum[v] * corr + ps;
      mrow[v] = mnew;
#pragma unroll
      for (int n = 0; n < 8; n++) accO[n][v] = accO[n][v] * corr;
      sP[wave][(v + 8 * lh) * 32 + ll]      = (bf16_t)p0;
      sP[wave][(v + 8 * lh) * 32 + 16 + ll] = (bf16_t)p1;
    }
    __syncthreads();

    // O += P * V   (A = 16x32 P, B = 32x16 V slices across HD)
    ABu up;
    up.q[0] = *(const u32x4*)&sP[wave][ll * 32 + lh * 8];
    up.q[1] = *(const u32x4*)&sP[wave][ll * 32 + lh * 8 + 16];
    v16bf pf = up.v;
#pragma unroll
    for (int n = 0; n < 8; n++) {
      ABu uv;
      uv.q[0] = *(const u32x4*)&sVt[(n * 16 + ll) * 32 + lh * 8];
      uv.q[1] = *(const u32x4*)&sVt[(n * 16 + ll) * 32 + lh * 8 + 16];
      v16bf bV = uv.v;
      accO[n] = wmma_bf16(pf, bV, accO[n]);
    }
    __syncthreads();
  }

  // epilogue: normalize and store bf16 in [b,s,h,d] layout
#pragma unroll
  for (int v = 0; v < 8; v++) {
    const float inv = 1.0f / lsum[v];
    const long qrow = qbase + v + 8 * lh;
#pragma unroll
    for (int n = 0; n < 8; n++)
      O[((long)(bb * S_) + qrow) * H_ + h * HD_ + n * 16 + ll] =
          (bf16_t)(accO[n][v] * inv);
  }
}

// ---------------- row-wise kernels (one 256-thread block per row, H=2048) ----
__global__ __launch_bounds__(256) void rmsnorm_bf_kernel(
    const float* __restrict__ x, const float* __restrict__ w,
    bf16_t* __restrict__ out)
{
  const long base = (long)blockIdx.x * H_;
  float xs[8], ss = 0.f;
#pragma unroll
  for (int k = 0; k < 8; k++) {
    xs[k] = x[base + threadIdx.x + k * 256];
    ss += xs[k] * xs[k];
  }
  ss = blk_sum(ss);
  const float r = rsqrtf(ss * (1.0f / H_) + EPS_);
#pragma unroll
  for (int k = 0; k < 8; k++) {
    const int c = threadIdx.x + k * 256;
    out[base + c] = (bf16_t)(xs[k] * r * w[c]);
  }
}

// X = X + rms(X, w_in); h_bf = rms(X_new, w_post)  (fused, one block per row)
__global__ __launch_bounds__(256) void rms_two_kernel(
    float* __restrict__ X, const float* __restrict__ win,
    const float* __restrict__ wpost, bf16_t* __restrict__ hb)
{
  const long base = (long)blockIdx.x * H_;
  float xs[8], ss = 0.f;
#pragma unroll
  for (int k = 0; k < 8; k++) {
    xs[k] = X[base + threadIdx.x + k * 256];
    ss += xs[k] * xs[k];
  }
  ss = blk_sum(ss);
  const float r1 = rsqrtf(ss * (1.0f / H_) + EPS_);
  float ss2 = 0.f;
#pragma unroll
  for (int k = 0; k < 8; k++) {
    const int c = threadIdx.x + k * 256;
    const float xn = xs[k] + xs[k] * r1 * win[c];
    xs[k] = xn;
    X[base + c] = xn;
    ss2 += xn * xn;
  }
  ss2 = blk_sum(ss2);
  const float r2 = rsqrtf(ss2 * (1.0f / H_) + EPS_);
#pragma unroll
  for (int k = 0; k < 8; k++) {
    const int c = threadIdx.x + k * 256;
    hb[base + c] = (bf16_t)(xs[k] * r2 * wpost[c]);
  }
}

// ---------------- elementwise kernels ----------------------------------------
__global__ __launch_bounds__(256) void f2bf_kernel(
    const float* __restrict__ in, bf16_t* __restrict__ out, long n)
{
  const long i = (long)blockIdx.x * 256 + threadIdx.x;
  if (i < n) out[i] = (bf16_t)in[i];
}

__global__ __launch_bounds__(256) void highway_out_kernel(
    const float* __restrict__ gatel, const float* __restrict__ mainl,
    const float* __restrict__ gb, float* __restrict__ out, long n)
{
  const long i = (long)blockIdx.x * 256 + threadIdx.x;
  if (i < n) {
    const int c = (int)(i & (H_ - 1));
    const float g = 1.0f / (1.0f + __expf(-(gatel[i] + gb[c])));
    out[i] = g * mainl[i];
  }
}

// X = x_mod + mod + sigmoid(gate + gb) * main
__global__ __launch_bounds__(256) void combine_x_kernel(
    const float* __restrict__ xmod, const float* __restrict__ mod,
    const float* __restrict__ gatel, const float* __restrict__ mainl,
    const float* __restrict__ gb, float* __restrict__ X, long n)
{
  const long i = (long)blockIdx.x * 256 + threadIdx.x;
  if (i < n) {
    const int c = (int)(i & (H_ - 1));
    const float g = 1.0f / (1.0f + __expf(-(gatel[i] + gb[c])));
    X[i] = xmod[i] + mod[i] + g * mainl[i];
  }
}

__global__ __launch_bounds__(256) void rope_kernel(
    const float* __restrict__ Qi, const float* __restrict__ Ki,
    bf16_t* __restrict__ Qo, bf16_t* __restrict__ Ko, long n)
{
  const long i = (long)blockIdx.x * 256 + threadIdx.x;
  if (i >= n) return;
  const int col = (int)(i & (H_ - 1));
  const long row = i >> 11;                 // / H_
  const int d = col & (HD_ - 1);
  const int s = (int)(row & (S_ - 1));
  const int j = d & 63;
  const float inv = __expf(-(float)j * ROPE_C);
  const float f = (float)s * inv;
  const float c = __cosf(f), sn = __sinf(f);
  const float rq = (d < 64) ? -Qi[i + 64] : Qi[i - 64];
  const float rk = (d < 64) ? -Ki[i + 64] : Ki[i - 64];
  Qo[i] = (bf16_t)(Qi[i] * c + rq * sn);
  Ko[i] = (bf16_t)(Ki[i] * c + rk * sn);
}

__global__ __launch_bounds__(256) void silu_mul_kernel(
    bf16_t* __restrict__ G, const bf16_t* __restrict__ U, long n)
{
  const long i = (long)blockIdx.x * 256 + threadIdx.x;
  if (i < n) {
    const float g = (float)G[i];
    const float u = (float)U[i];
    G[i] = (bf16_t)((g / (1.0f + __expf(-g))) * u);
  }
}

__global__ __launch_bounds__(256) void add_out_kernel(
    const float* __restrict__ X, const float* __restrict__ mlp,
    float* __restrict__ out, long n)
{
  const long i = (long)blockIdx.x * 256 + threadIdx.x;
  if (i < n) out[i] = X[i] + mlp[i];
}

// ---------------- host orchestration -----------------------------------------
static inline long cdiv(long a, long b) { return (a + b - 1) / b; }

extern "C" void kernel_launch(void* const* d_in, const int* in_sizes, int n_in,
                              void* d_out, int out_size, void* d_ws, size_t ws_size,
                              hipStream_t stream) {
  (void)in_sizes; (void)n_in; (void)out_size; (void)ws_size;
  const float* x_mod      = (const float*)d_in[0];
  const float* x_back     = (const float*)d_in[1];
  const float* mask       = (const float*)d_in[2];
  const float* wq         = (const float*)d_in[3];
  const float* wk         = (const float*)d_in[4];
  const float* wv         = (const float*)d_in[5];
  const float* wo         = (const float*)d_in[6];
  const float* cross_ln_w = (const float*)d_in[7];
  const float* fb_w       = (const float*)d_in[8];
  const float* fb_g       = (const float*)d_in[9];
  const float* fb_gb      = (const float*)d_in[10];
  const float* dr_w       = (const float*)d_in[11];
  const float* dr_g       = (const float*)d_in[12];
  const float* dr_gb      = (const float*)d_in[13];
  const float* in_ln_w    = (const float*)d_in[14];
  const float* post_ln_w  = (const float*)d_in[15];
  const float* w_gate     = (const float*)d_in[16];
  const float* w_up       = (const float*)d_in[17];
  const float* w_down     = (const float*)d_in[18];

  float* out_x     = (float*)d_out;
  float* out_delta = out_x + (long)NROW * H_;

  const long HH  = (long)H_ * H_;        // 4.19M
  const long FH  = (long)FF_ * H_;       // 16.8M
  const long NHn = (long)NROW * H_;      // 8.39M
  const long NFn = (long)NROW * FF_;     // 33.6M

  char* p = (char*)d_ws;
  auto alloc = [&](size_t bytes) -> void* {
    void* r = (void*)p;
    p += (bytes + 255) & ~(size_t)255;
    return r;
  };
  bf16_t* wq_b    = (bf16_t*)alloc(HH * 2);
  bf16_t* wk_b    = (bf16_t*)alloc(HH * 2);
  bf16_t* wv_b    = (bf16_t*)alloc(HH * 2);
  bf16_t* wo_b    = (bf16_t*)alloc(HH * 2);
  bf16_t* fbw_b   = (bf16_t*)alloc(HH * 2);
  bf16_t* fbg_b   = (bf16_t*)alloc(HH * 2);
  bf16_t* drw_b   = (bf16_t*)alloc(HH * 2);
  bf16_t* drg_b   = (bf16_t*)alloc(HH * 2);
  bf16_t* wgate_b = (bf16_t*)alloc(FH * 2);
  bf16_t* wup_b   = (bf16_t*)alloc(FH * 2);
  bf16_t* wdown_b = (bf16_t*)alloc(FH * 2);
  bf16_t* xmod_b  = (bf16_t*)alloc(NHn * 2);
  bf16_t* xback_b = (bf16_t*)alloc(NHn * 2);
  bf16_t* qn_b    = (bf16_t*)alloc(NHn * 2);
  bf16_t* h_b     = (bf16_t*)alloc(NHn * 2);
  bf16_t* Qb      = (bf16_t*)alloc(NHn * 2);
  bf16_t* Kb      = (bf16_t*)alloc(NHn * 2);
  bf16_t* Vb      = (bf16_t*)alloc(NHn * 2);
  bf16_t* Ob      = (bf16_t*)alloc(NHn * 2);
  float*  t0      = (float*) alloc(NHn * 4);
  float*  t1      = (float*) alloc(NHn * 4);
  float*  t2      = (float*) alloc(NHn * 4);
  float*  Xf      = (float*) alloc(NHn * 4);
  bf16_t* Gb      = (bf16_t*)alloc(NFn * 2);
  bf16_t* Ub      = (bf16_t*)alloc(NFn * 2);

  auto cvt = [&](const float* src, bf16_t* dst, long n) {
    f2bf_kernel<<<dim3((unsigned)cdiv(n, 256)), 256, 0, stream>>>(src, dst, n);
  };
  auto gemm = [&](const bf16_t* A, const bf16_t* W, void* C,
                  int M, int Nc, int K, int bf) {
    gemm_bt_kernel<<<dim3(Nc / 128, M / 128), 256, 0, stream>>>(A, W, C, M, Nc, K, bf);
  };

  // --- precision conversion ---
  cvt(wq, wq_b, HH);   cvt(wk, wk_b, HH);   cvt(wv, wv_b, HH);   cvt(wo, wo_b, HH);
  cvt(fb_w, fbw_b, HH); cvt(fb_g, fbg_b, HH);
  cvt(dr_w, drw_b, HH); cvt(dr_g, drg_b, HH);
  cvt(w_gate, wgate_b, FH); cvt(w_up, wup_b, FH); cvt(w_down, wdown_b, FH);
  cvt(x_mod, xmod_b, NHn);  cvt(x_back, xback_b, NHn);

  // --- feedback highway -> delta (second output) ---
  gemm(xmod_b, fbg_b, t1, NROW, H_, H_, 0);
  gemm(xmod_b, fbw_b, t2, NROW, H_, H_, 0);
  highway_out_kernel<<<dim3((unsigned)cdiv(NHn, 256)), 256, 0, stream>>>(
      t1, t2, fb_gb, out_delta, NHn);

  // --- cross attention ---
  rmsnorm_bf_kernel<<<dim3(NROW), 256, 0, stream>>>(x_mod, cross_ln_w, qn_b);
  gemm(qn_b,    wq_b, t0, NROW, H_, H_, 0);   // Q (f32)
  gemm(xback_b, wk_b, t1, NROW, H_, H_, 0);   // K (f32)
  rope_kernel<<<dim3((unsigned)cdiv(NHn, 256)), 256, 0, stream>>>(
      t0, t1, Qb, Kb, NHn);
  gemm(xback_b, wv_b, Vb, NROW, H_, H_, 1);   // V (bf16)
  attn_kernel<<<dim3(S_ / 128, NH_, B_), 256, 0, stream>>>(Qb, Kb, Vb, mask, Ob);
  gemm(Ob, wo_b, t0, NROW, H_, H_, 0);        // mod (f32)

  // --- driver highway + combine ---
  gemm(xback_b, drg_b, t1, NROW, H_, H_, 0);
  gemm(xback_b, drw_b, t2, NROW, H_, H_, 0);
  combine_x_kernel<<<dim3((unsigned)cdiv(NHn, 256)), 256, 0, stream>>>(
      x_mod, t0, t1, t2, dr_gb, Xf, NHn);

  // --- decoder: x += rms(x,in); h = rms(x,post) ---
  rms_two_kernel<<<dim3(NROW), 256, 0, stream>>>(Xf, in_ln_w, post_ln_w, h_b);

  // --- SwiGLU MLP ---
  gemm(h_b, wgate_b, Gb, NROW, FF_, H_, 1);
  gemm(h_b, wup_b,   Ub, NROW, FF_, H_, 1);
  silu_mul_kernel<<<dim3((unsigned)cdiv(NFn, 256)), 256, 0, stream>>>(Gb, Ub, NFn);
  gemm(Gb, wdown_b, t0, NROW, H_, FF_, 0);

  // --- final residual -> first output ---
  add_out_kernel<<<dim3((unsigned)cdiv(NHn, 256)), 256, 0, stream>>>(
      Xf, t0, out_x, NHn);
}
